// PaperInformedMambaLMHeadModel_23785528885644
// MI455X (gfx1250) — matrix operand order repaired and checked by reading, not compiled, not run on hardware
//
#include <hip/hip_runtime.h>
#include <hip/hip_bf16.h>
#include <math.h>

// ---- model dims ----
#define DM   512      // d_model
#define NS   32       // d_state
#define VOC  32000
#define NLAY 4
#define LSEQ 2048
#define KC   4        // conv kernel

typedef _Float16 v16h __attribute__((ext_vector_type(16)));
typedef _Float16 v8h  __attribute__((ext_vector_type(8)));
typedef _Float16 v4h  __attribute__((ext_vector_type(4)));
typedef float    v8f  __attribute__((ext_vector_type(8)));

// =====================================================================
// Tiled WMMA GEMM: C[M,N] = A[M,K] * W[N,K]^T (+bias) (+softplus/clamp)
// Block: 256 threads (8 waves). Tile: BM=128, BN=128, BK=32, double-
// buffered LDS. Wave grid 2x4; wave tile 64x32 = 4x2 wmma 16x16 tiles.
// MODE: 0 = +bias, 1 = +bias, softplus, clamp[1e-3,0.2], 2 = plain
// =====================================================================
template <int MODE>
__global__ __launch_bounds__(256) void gemm_wmma_f16(
    const float* __restrict__ A, const float* __restrict__ W,
    const float* __restrict__ bias, float* __restrict__ C,
    int M, int Nout, int Kd)
{
    __shared__ __align__(16) _Float16 As[2 * 128 * 48];   // 2 x (128 rows x 32 K, pad->48)
    __shared__ __align__(16) _Float16 Bs[2 * 128 * 48];

    const int t    = threadIdx.x;
    const int lane = t & 31;
    const int wv   = t >> 5;
    const int wm   = wv >> 2;       // 0..1  (64 M-rows each)
    const int wn   = wv & 3;        // 0..3  (32 N-cols each)
    const int g    = lane >> 4;     // half-wave group
    const int mn   = lane & 15;
    const int mBase = blockIdx.y * 128;
    const int nBase = blockIdx.x * 128;
    const int nk = Kd >> 5;

    // staging chunk coordinates (4 float4 chunks per thread for each tile)
    int rows[4], cols[4];
    #pragma unroll
    for (int i = 0; i < 4; ++i) { int c = t + i * 256; rows[i] = c >> 3; cols[i] = (c & 7) * 4; }

    float4 areg[4], breg[4];

    // ---- preload tile 0: issue ALL global loads, then convert+store ----
    #pragma unroll
    for (int i = 0; i < 4; ++i)
        areg[i] = *(const float4*)(A + (size_t)(mBase + rows[i]) * Kd + cols[i]);
    #pragma unroll
    for (int i = 0; i < 4; ++i)
        breg[i] = *(const float4*)(W + (size_t)(nBase + rows[i]) * Kd + cols[i]);
    #pragma unroll
    for (int i = 0; i < 4; ++i) {
        v4h ha = { (_Float16)areg[i].x, (_Float16)areg[i].y, (_Float16)areg[i].z, (_Float16)areg[i].w };
        *(v4h*)(As + rows[i] * 48 + cols[i]) = ha;
        v4h hb = { (_Float16)breg[i].x, (_Float16)breg[i].y, (_Float16)breg[i].z, (_Float16)breg[i].w };
        *(v4h*)(Bs + rows[i] * 48 + cols[i]) = hb;
    }
    __syncthreads();

    v8f acc[4][2] = {};

    for (int kt = 0; kt < nk; ++kt) {
        const int cur = kt & 1, nxt = cur ^ 1;
        const _Float16* AsC = As + cur * (128 * 48);
        const _Float16* BsC = Bs + cur * (128 * 48);

        // ---- issue next tile's global loads (overlap with wmma below) ----
        if (kt + 1 < nk) {
            const int k0 = (kt + 1) << 5;
            #pragma unroll
            for (int i = 0; i < 4; ++i)
                areg[i] = *(const float4*)(A + (size_t)(mBase + rows[i]) * Kd + k0 + cols[i]);
            #pragma unroll
            for (int i = 0; i < 4; ++i)
                breg[i] = *(const float4*)(W + (size_t)(nBase + rows[i]) * Kd + k0 + cols[i]);
        }

        // ---- fragments (per ISA 16-bit A / B VGPR striping) ----
        v16h af[4], bf[2];
        #pragma unroll
        for (int mi = 0; mi < 4; ++mi) {
            int row = wm * 64 + mi * 16 + mn;
            v8h lo = *(const v8h*)(AsC + row * 48 + 8 * g);        // K = 8g..8g+7
            v8h hi = *(const v8h*)(AsC + row * 48 + 16 + 8 * g);   // K = 16+8g..+7
            af[mi] = __builtin_shufflevector(lo, hi,
                     0,1,2,3,4,5,6,7,8,9,10,11,12,13,14,15);
        }
        #pragma unroll
        for (int ni = 0; ni < 2; ++ni) {
            int row = wn * 32 + ni * 16 + mn;                      // N index
            v8h lo = *(const v8h*)(BsC + row * 48 + 8 * g);        // K = 8g..8g+7
            v8h hi = *(const v8h*)(BsC + row * 48 + 16 + 8 * g);   // K = 16+8g..+7
            bf[ni] = __builtin_shufflevector(lo, hi,
                     0,1,2,3,4,5,6,7,8,9,10,11,12,13,14,15);
        }
        #pragma unroll
        for (int mi = 0; mi < 4; ++mi)
            #pragma unroll
            for (int ni = 0; ni < 2; ++ni)
                acc[mi][ni] = __builtin_amdgcn_wmma_f32_16x16x32_f16(
                    false, af[mi], false, bf[ni], (short)0, acc[mi][ni], false, false);

        // ---- convert + store next tile into the other LDS buffer ----
        if (kt + 1 < nk) {
            _Float16* AsN = As + nxt * (128 * 48);
            _Float16* BsN = Bs + nxt * (128 * 48);
            #pragma unroll
            for (int i = 0; i < 4; ++i) {
                v4h ha = { (_Float16)areg[i].x, (_Float16)areg[i].y, (_Float16)areg[i].z, (_Float16)areg[i].w };
                *(v4h*)(AsN + rows[i] * 48 + cols[i]) = ha;
                v4h hb = { (_Float16)breg[i].x, (_Float16)breg[i].y, (_Float16)breg[i].z, (_Float16)breg[i].w };
                *(v4h*)(BsN + rows[i] * 48 + cols[i]) = hb;
            }
        }
        __syncthreads();
    }

    // ---- epilogue: C/D layout -> row = base + r + 8g, col = base + (lane%16) ----
    #pragma unroll
    for (int mi = 0; mi < 4; ++mi) {
        #pragma unroll
        for (int ni = 0; ni < 2; ++ni) {
            int rowBase = mBase + wm * 64 + mi * 16 + 8 * g;
            int col     = nBase + wn * 32 + ni * 16 + mn;
            float bv = (MODE == 2) ? 0.f : bias[col];
            #pragma unroll
            for (int r = 0; r < 8; ++r) {
                float v = acc[mi][ni][r] + bv;
                if (MODE == 1) {
                    v = (v > 20.f) ? v : log1pf(expf(v));   // softplus
                    v = fminf(fmaxf(v, 0.001f), 0.2f);      // clamp [DT_MIN, DT_MAX]
                }
                C[(size_t)(rowBase + r) * Nout + col] = v;
            }
        }
    }
}

// =====================================================================
// Embedding gather: x[l,d] = emb[ids[l], d]
// =====================================================================
__global__ __launch_bounds__(256) void embed_kernel(
    const float* __restrict__ emb, const int* __restrict__ ids,
    float* __restrict__ x)
{
    int idx = blockIdx.x * 256 + threadIdx.x;       // [0, L*D)
    int l = idx >> 9, d = idx & (DM - 1);
    x[idx] = emb[(size_t)ids[l] * DM + d];
}

// =====================================================================
// Depthwise causal conv (K=4, left pad 3) + bias + SiLU.
// input xc = xz[:, 0:512]
// =====================================================================
__global__ __launch_bounds__(256) void dwconv_silu_kernel(
    const float* __restrict__ xz, const float* __restrict__ cw,
    const float* __restrict__ cb, float* __restrict__ x2)
{
    int idx = blockIdx.x * 256 + threadIdx.x;
    int l = idx >> 9, d = idx & (DM - 1);
    float acc = cb[d];
    #pragma unroll
    for (int k = 0; k < KC; ++k) {
        int ls = l - (KC - 1) + k;
        if (ls >= 0) acc += cw[d * KC + k] * xz[(size_t)ls * (2 * DM) + d];
    }
    x2[idx] = acc / (1.f + __expf(-acc));           // silu
}

// =====================================================================
// SSM scan. One wave per channel d, lane = state n (N=32 == wave32).
// Software-pipelined: next step's x/dt loads issue before current math.
// Partials buffered 32 steps in LDS, transpose-reduced off the critical
// path. Output fused with gate: y = sum * silu(z).
// =====================================================================
__global__ __launch_bounds__(256) void ssm_scan_kernel(
    const float* __restrict__ x2, const float* __restrict__ dtv,
    const float* __restrict__ xz, const float* __restrict__ A_log,
    const float* __restrict__ Bw, const float* __restrict__ Bb,
    const float* __restrict__ Cw, const float* __restrict__ Cb,
    float* __restrict__ y)
{
    __shared__ __align__(16) float pbuf[8][32 * 36];   // per-wave 32 steps x 32 lanes (pad 36)
    const int lane = threadIdx.x & 31;
    const int wv   = threadIdx.x >> 5;
    const int d    = blockIdx.x * 8 + wv;

    const float a   = -expf(fminf(fmaxf(A_log[d * NS + lane], -5.f), 2.f));
    const float bw  = Bw[d * NS + lane], bb = Bb[d * NS + lane];
    const float cwv = Cw[d * NS + lane], cbv = Cb[d * NS + lane];
    float s = 0.f;
    float* pw = pbuf[wv];

    for (int c = 0; c < LSEQ / 32; ++c) {
        const int lbase = c * 32;
        float xv = x2[(size_t)lbase * DM + d];
        float dt = dtv[(size_t)lbase * DM + d];
        #pragma unroll 8
        for (int j = 0; j < 32; ++j) {
            float xn = 0.f, dn = 0.f;
            if (j < 31) {                                  // prefetch next step
                xn = x2[(size_t)(lbase + j + 1) * DM + d];
                dn = dtv[(size_t)(lbase + j + 1) * DM + d];
            }
            float abar = __expf(fminf(fmaxf(dt * a, -10.f), 0.f));
            s = fminf(fmaxf(s * abar + dt * (xv * bw + bb) * xv, -10.f), 10.f);
            pw[j * 36 + lane] = (xv * cwv + cbv) * s;
            xv = xn; dt = dn;
        }
        __syncthreads();
        // transpose-reduce: lane j sums its 32 partials (8 x float4 LDS loads)
        float sum = 0.f;
        const float4* pr = (const float4*)(pw + lane * 36);
        #pragma unroll
        for (int q = 0; q < 8; ++q) { float4 f = pr[q]; sum += f.x + f.y + f.z + f.w; }
        int l = lbase + lane;
        float z = xz[(size_t)l * (2 * DM) + DM + d];
        y[(size_t)l * DM + d] = sum * (z / (1.f + __expf(-z)));   // * silu(z)
        __syncthreads();
    }
}

// =====================================================================
// residual add + LayerNorm (one wave per row, wave32 shfl_xor reduce)
// x <- LN(tmp + x) * g + b    (tmp already contains out_b)
// =====================================================================
__global__ __launch_bounds__(256) void add_ln_kernel(
    const float* __restrict__ tmp, float* __restrict__ x,
    const float* __restrict__ gm, const float* __restrict__ bt)
{
    const int lane = threadIdx.x & 31;
    const int wv   = threadIdx.x >> 5;
    const int row  = blockIdx.x * 8 + wv;
    float vals[16], sum = 0.f, sq = 0.f;
    #pragma unroll
    for (int i = 0; i < 16; ++i) {
        int cidx = i * 32 + lane;
        float v = tmp[(size_t)row * DM + cidx] + x[(size_t)row * DM + cidx];
        vals[i] = v; sum += v; sq += v * v;
    }
    #pragma unroll
    for (int off = 16; off >= 1; off >>= 1) {
        sum += __shfl_xor(sum, off, 32);
        sq  += __shfl_xor(sq, off, 32);
    }
    float mean = sum * (1.f / DM);
    float var  = sq * (1.f / DM) - mean * mean;
    float inv  = rsqrtf(var + 1e-5f);
    #pragma unroll
    for (int i = 0; i < 16; ++i) {
        int cidx = i * 32 + lane;
        x[(size_t)row * DM + cidx] = (vals[i] - mean) * inv * gm[cidx] + bt[cidx];
    }
}

// =====================================================================
// host launcher
// =====================================================================
extern "C" void kernel_launch(void* const* d_in, const int* in_sizes, int n_in,
                              void* d_out, int out_size, void* d_ws, size_t ws_size,
                              hipStream_t stream)
{
    const float* emb   = (const float*)d_in[0];
    const float* inW   = (const float*)d_in[1];
    const float* inB   = (const float*)d_in[2];
    const float* cW    = (const float*)d_in[3];
    const float* cB    = (const float*)d_in[4];
    const float* Alog  = (const float*)d_in[5];
    const float* BpW   = (const float*)d_in[6];
    const float* BpB   = (const float*)d_in[7];
    const float* CpW   = (const float*)d_in[8];
    const float* CpB   = (const float*)d_in[9];
    const float* dtW   = (const float*)d_in[10];
    const float* dtB   = (const float*)d_in[11];
    const float* oW    = (const float*)d_in[12];
    const float* oB    = (const float*)d_in[13];
    const float* lnG   = (const float*)d_in[14];
    const float* lnB   = (const float*)d_in[15];
    const float* lmW   = (const float*)d_in[16];
    const int*   ids   = (const int*)d_in[17];
    float* out = (float*)d_out;

    const size_t LD = (size_t)LSEQ * DM;      // 1,048,576 floats
    float* ws  = (float*)d_ws;
    float* x   = ws;                          // [L,D]
    float* xz  = x  + LD;                     // [L,2D]
    float* x2  = xz + 2 * LD;                 // [L,D]
    float* dtv = x2 + LD;                     // [L,D]
    float* yv  = dtv + LD;                    // [L,D]
    float* tmp = yv + LD;                     // [L,D]

    embed_kernel<<<(LSEQ * DM) / 256, 256, 0, stream>>>(emb, ids, x);

    for (int l = 0; l < NLAY; ++l) {
        // xz = x @ in_w.T + in_b                     [2048,1024]
        gemm_wmma_f16<0><<<dim3(2 * DM / 128, LSEQ / 128), 256, 0, stream>>>(
            x, inW + (size_t)l * 2 * DM * DM, inB + (size_t)l * 2 * DM,
            xz, LSEQ, 2 * DM, DM);
        // x2 = silu(depthwise_conv(xc) + cb)
        dwconv_silu_kernel<<<(LSEQ * DM) / 256, 256, 0, stream>>>(
            xz, cW + (size_t)l * DM * KC, cB + (size_t)l * DM, x2);
        // dt = clamp(softplus(x2 @ dt_w.T + dt_b))
        gemm_wmma_f16<1><<<dim3(DM / 128, LSEQ / 128), 256, 0, stream>>>(
            x2, dtW + (size_t)l * DM * DM, dtB + (size_t)l * DM,
            dtv, LSEQ, DM, DM);
        // sequential SSM scan + gate with silu(z)
        ssm_scan_kernel<<<DM / 8, 256, 0, stream>>>(
            x2, dtv, xz,
            Alog + (size_t)l * DM * NS,
            BpW + (size_t)l * DM * NS, BpB + (size_t)l * DM * NS,
            CpW + (size_t)l * DM * NS, CpB + (size_t)l * DM * NS, yv);
        // tmp = y @ out_w.T + out_b
        gemm_wmma_f16<0><<<dim3(DM / 128, LSEQ / 128), 256, 0, stream>>>(
            yv, oW + (size_t)l * DM * DM, oB + (size_t)l * DM,
            tmp, LSEQ, DM, DM);
        // x = LN(tmp + x)
        add_ln_kernel<<<LSEQ / 8, 256, 0, stream>>>(
            tmp, x, lnG + (size_t)l * DM, lnB + (size_t)l * DM);
    }

    // logits = x @ lm_head.T                         [2048,32000]
    gemm_wmma_f16<2><<<dim3(VOC / 128, LSEQ / 128), 256, 0, stream>>>(
        x, lmW, nullptr, out, LSEQ, VOC, DM);
}
